// TwinSAGE_34548716929225
// MI455X (gfx1250) — compile-verified, matrix operand order checked by reading, not run on hardware
//
#include <hip/hip_runtime.h>

typedef __attribute__((ext_vector_type(16))) _Float16 v16h;
typedef __attribute__((ext_vector_type(8)))  float    v8f;

static constexpr int NN   = 50000;   // nodes (== 3125 * 16, exact tiles)
static constexpr int EE   = 800000;  // edges
static constexpr int FIN  = 128;     // input features
static constexpr int HID  = 64;      // hidden
static constexpr int CLS  = 40;      // classes
static constexpr int CLSP = 48;      // classes padded to 16-col tiles

// ---------------------------------------------------------------- utilities
__global__ void k_zero(float* __restrict__ p, int n) {
    int i = blockIdx.x * blockDim.x + threadIdx.x;
    if (i < n) p[i] = 0.0f;
}

// Pack one f32 weight matrix (K x srcCols, row-major) into WMMA-fragment-major
// f16 layout: [kchunk][coltile][lane][16 halves], so a B fragment is a single
// contiguous 32-byte (v16h) load per lane. cols >= srcCols are zero padding.
__device__ __forceinline__ void pack_frag(const float* __restrict__ W,
                                          _Float16* __restrict__ Wp,
                                          int idx, int Nout, int srcCols) {
    const int i      = idx & 15;
    const int lane   = (idx >> 4) & 31;
    const int rest   = idx >> 9;
    const int tilesN = Nout / 16;
    const int t      = rest % tilesN;
    const int c      = rest / tilesN;                 // k-chunk (32 wide)
    const int k      = c * 32 + ((lane >> 4) << 4) + i;
    const int col    = t * 16 + (lane & 15);
    const float v    = (col < srcCols) ? W[(size_t)k * srcCols + col] : 0.0f;
    Wp[idx] = (_Float16)v;
}

__global__ void k_cvt_weights(const float* __restrict__ Wl0, const float* __restrict__ Wr0,
                              const float* __restrict__ Wl1, const float* __restrict__ Wr1,
                              const float* __restrict__ outW,
                              _Float16* __restrict__ Wl0p, _Float16* __restrict__ Wr0p,
                              _Float16* __restrict__ Wl1p, _Float16* __restrict__ Wr1p,
                              _Float16* __restrict__ outWp) {
    int i = blockIdx.x * blockDim.x + threadIdx.x;
    const int n0 = FIN * HID;          // 8192
    const int n1 = 2 * n0;             // 16384
    const int n2 = n1 + HID * HID;     // 20480
    const int n3 = n2 + HID * HID;     // 24576
    const int n4 = n3 + HID * CLSP;    // 27648
    if (i < n0)       pack_frag(Wl0,  Wl0p, i,       HID,  HID);
    else if (i < n1)  pack_frag(Wr0,  Wr0p, i - n0,  HID,  HID);
    else if (i < n2)  pack_frag(Wl1,  Wl1p, i - n1,  HID,  HID);
    else if (i < n3)  pack_frag(Wr1,  Wr1p, i - n2,  HID,  HID);
    else if (i < n4)  pack_frag(outW, outWp, i - n3, CLSP, CLS);
}

__global__ void k_deg(const int* __restrict__ dst, float* __restrict__ deg) {
    int e = blockIdx.x * blockDim.x + threadIdx.x;
    if (e < EE) atomicAdd(&deg[dst[e]], 1.0f);
}

__global__ void k_invdeg(float* __restrict__ deg) {
    int n = blockIdx.x * blockDim.x + threadIdx.x;
    if (n < NN) deg[n] = 1.0f / fmaxf(deg[n], 1.0f);
}

// One block per edge, F threads: agg[dst] += X[src]   (global_atomic_add_f32)
__global__ void k_scatter(const float* __restrict__ X, const int* __restrict__ src,
                          const int* __restrict__ dst, float* __restrict__ agg, int F) {
    int e = blockIdx.x;
    int f = threadIdx.x;
    atomicAdd(&agg[(size_t)dst[e] * F + f], X[(size_t)src[e] * F + f]);
}

// ---------------------------------------------------------------- WMMA frags
// A fragment (16x32, f16): lane&15 = row, lane>=16 -> K upper half of chunk.
__device__ __forceinline__ v16h afrag(const float* __restrict__ A, int lda,
                                      int grow, int kofs, float s) {
    const float* p = A + (size_t)grow * lda + kofs;
    v16h a;
#pragma unroll
    for (int i = 0; i < 16; ++i) a[i] = (_Float16)(p[i] * s);
    return a;
}

// B fragment from fragment-major packed weights: one 32-byte vector load.
__device__ __forceinline__ v16h bfrag(const _Float16* __restrict__ Wp, int tilesN,
                                      int kc, int t, int lane) {
    return *(const v16h*)(Wp + (((size_t)(kc >> 5) * tilesN + t) << 9) + (lane << 4));
}

// H = relu( (Agg * invdeg) @ Wl + X @ Wr + bias )   -- Nout = 64, one wave/tile
__global__ __launch_bounds__(32)
void k_sage_gemm(const float* __restrict__ Agg, const float* __restrict__ invdeg,
                 const float* __restrict__ X,
                 const _Float16* __restrict__ Wlp, const _Float16* __restrict__ Wrp,
                 const float* __restrict__ bias, float* __restrict__ Out, int K) {
    const int lane = threadIdx.x & 31;
    const int r15  = lane & 15;
    const int ksel = (lane >> 4) << 4;            // 0 or 16
    const int grow = blockIdx.x * 16 + r15;
    const float sc = invdeg[grow];

    v8f acc[4] = {};
    for (int kc = 0; kc < K; kc += 32) {
        if (kc + 32 < K)
            __builtin_prefetch(Agg + (size_t)grow * K + kc + 32, 0, 3);
        v16h aA = afrag(Agg, K, grow, kc + ksel, sc);
        v16h aX = afrag(X,   K, grow, kc + ksel, 1.0f);
#pragma unroll
        for (int t = 0; t < 4; ++t) {
            v16h bL = bfrag(Wlp, 4, kc, t, lane);
            v16h bR = bfrag(Wrp, 4, kc, t, lane);
            acc[t] = __builtin_amdgcn_wmma_f32_16x16x32_f16(false, aA, false, bL,
                                                            (short)0, acc[t], false, false);
            acc[t] = __builtin_amdgcn_wmma_f32_16x16x32_f16(false, aX, false, bR,
                                                            (short)0, acc[t], false, false);
        }
    }
    const int mofs = (lane >> 4) << 3;            // 0 or 8
#pragma unroll
    for (int t = 0; t < 4; ++t) {
        const int col = t * 16 + r15;
        const float bv = bias[col];
#pragma unroll
        for (int j = 0; j < 8; ++j) {
            float v = fmaxf(acc[t][j] + bv, 0.0f);
            Out[(size_t)(blockIdx.x * 16 + mofs + j) * HID + col] = v;
        }
    }
}

// Ht = relu( Xt @ Wr )   -- Nout = 64
__global__ __launch_bounds__(32)
void k_twin_gemm(const float* __restrict__ Xt, const _Float16* __restrict__ Wrp,
                 float* __restrict__ Out, int K) {
    const int lane = threadIdx.x & 31;
    const int r15  = lane & 15;
    const int ksel = (lane >> 4) << 4;
    const int grow = blockIdx.x * 16 + r15;

    v8f acc[4] = {};
    for (int kc = 0; kc < K; kc += 32) {
        v16h aX = afrag(Xt, K, grow, kc + ksel, 1.0f);
#pragma unroll
        for (int t = 0; t < 4; ++t) {
            v16h bR = bfrag(Wrp, 4, kc, t, lane);
            acc[t] = __builtin_amdgcn_wmma_f32_16x16x32_f16(false, aX, false, bR,
                                                            (short)0, acc[t], false, false);
        }
    }
    const int mofs = (lane >> 4) << 3;
#pragma unroll
    for (int t = 0; t < 4; ++t) {
        const int col = t * 16 + r15;
#pragma unroll
        for (int j = 0; j < 8; ++j)
            Out[(size_t)(blockIdx.x * 16 + mofs + j) * HID + col] = fmaxf(acc[t][j], 0.0f);
    }
}

// One wave32 per node: lanes cover features d and d+32 (coalesced), dot
// products reduced with shfl_xor, then alpha softmax + blended hsum.
__global__ __launch_bounds__(256)
void k_attn(const float* __restrict__ h0, const float* __restrict__ h0t,
            const float* __restrict__ h1, const float* __restrict__ h1t,
            float* __restrict__ hsum, float* __restrict__ alpha) {
    const int lane = threadIdx.x & 31;
    const int n = blockIdx.x * 8 + (threadIdx.x >> 5);
    if (n >= NN) return;                       // uniform per wave
    const float* p0  = h0  + (size_t)n * HID;
    const float* p0t = h0t + (size_t)n * HID;
    const float* p1  = h1  + (size_t)n * HID;
    const float* p1t = h1t + (size_t)n * HID;

    const float x0a = p0[lane],      x0b = p0[lane + 32];
    const float x1a = p1[lane],      x1b = p1[lane + 32];
    float s0 = x0a * p0t[lane] + x0b * p0t[lane + 32];
    float s1 = x1a * p1t[lane] + x1b * p1t[lane + 32];
#pragma unroll
    for (int o = 16; o > 0; o >>= 1) {
        s0 += __shfl_xor(s0, o, 32);
        s1 += __shfl_xor(s1, o, 32);
    }
    // TEMP = 1.0
    const float mx  = fmaxf(s0, s1);
    const float e0  = __expf(s0 - mx);
    const float e1  = __expf(s1 - mx);
    const float inv = 1.0f / (e0 + e1);
    const float a0  = e0 * inv, a1 = e1 * inv;
    if (lane == 0) {
        alpha[(size_t)n * 2 + 0] = a0;
        alpha[(size_t)n * 2 + 1] = a1;
    }
    float* hs = hsum + (size_t)n * HID;
    hs[lane]      = a0 * x0a + a1 * x1a;
    hs[lane + 32] = a0 * x0b + a1 * x1b;
}

// Out[N,40] = Hsum @ outWp(64x48 packed) + outb  (store only cols < 40)
__global__ __launch_bounds__(32)
void k_head(const float* __restrict__ H, const _Float16* __restrict__ Wp,
            const float* __restrict__ outb, float* __restrict__ Out) {
    const int lane = threadIdx.x & 31;
    const int r15  = lane & 15;
    const int ksel = (lane >> 4) << 4;
    const int grow = blockIdx.x * 16 + r15;

    v8f acc[3] = {};
    for (int kc = 0; kc < HID; kc += 32) {
        v16h aH = afrag(H, HID, grow, kc + ksel, 1.0f);
#pragma unroll
        for (int t = 0; t < 3; ++t) {
            v16h b = bfrag(Wp, 3, kc, t, lane);
            acc[t] = __builtin_amdgcn_wmma_f32_16x16x32_f16(false, aH, false, b,
                                                            (short)0, acc[t], false, false);
        }
    }
    const int mofs = (lane >> 4) << 3;
#pragma unroll
    for (int t = 0; t < 3; ++t) {
        const int col = t * 16 + r15;
        if (col < CLS) {
            const float bv = outb[col];
#pragma unroll
            for (int j = 0; j < 8; ++j)
                Out[(size_t)(blockIdx.x * 16 + mofs + j) * CLS + col] = acc[t][j] + bv;
        }
    }
}

// ---------------------------------------------------------------- launcher
extern "C" void kernel_launch(void* const* d_in, const int* in_sizes, int n_in,
                              void* d_out, int out_size, void* d_ws, size_t ws_size,
                              hipStream_t stream) {
    (void)in_sizes; (void)n_in; (void)out_size; (void)ws_size;

    const float* x    = (const float*)d_in[0];
    const int*   ei   = (const int*)d_in[1];
    const int*   src  = ei;
    const int*   dst  = ei + EE;
    const float* Wl0  = (const float*)d_in[2];
    const float* Wr0  = (const float*)d_in[3];
    const float* b0   = (const float*)d_in[4];
    const float* Wl1  = (const float*)d_in[5];
    const float* Wr1  = (const float*)d_in[6];
    const float* b1   = (const float*)d_in[7];
    const float* outW = (const float*)d_in[8];
    const float* outb = (const float*)d_in[9];
    float* out = (float*)d_out;

    // workspace layout (deg, agg0, agg1 contiguous so one zero-fill covers them)
    float* deg  = (float*)d_ws;
    float* agg0 = deg  + NN;
    float* agg1 = agg0 + (size_t)NN * FIN;
    float* h0   = agg1 + (size_t)NN * HID;
    float* h0t  = h0   + (size_t)NN * HID;
    float* h1   = h0t  + (size_t)NN * HID;
    float* h1t  = h1   + (size_t)NN * HID;
    float* hsum = h1t  + (size_t)NN * HID;
    _Float16* Wl0p  = (_Float16*)(hsum + (size_t)NN * HID);
    _Float16* Wr0p  = Wl0p + FIN * HID;
    _Float16* Wl1p  = Wr0p + FIN * HID;
    _Float16* Wr1p  = Wl1p + HID * HID;
    _Float16* outWp = Wr1p + HID * HID;

    const int zn = NN * (1 + FIN + HID);
    k_zero<<<(zn + 255) / 256, 256, 0, stream>>>(deg, zn);

    const int wn = 2 * FIN * HID + 2 * HID * HID + HID * CLSP;
    k_cvt_weights<<<(wn + 255) / 256, 256, 0, stream>>>(Wl0, Wr0, Wl1, Wr1, outW,
                                                        Wl0p, Wr0p, Wl1p, Wr1p, outWp);

    k_deg<<<(EE + 255) / 256, 256, 0, stream>>>(dst, deg);
    k_invdeg<<<(NN + 255) / 256, 256, 0, stream>>>(deg);

    // layer 0
    k_scatter<<<EE, FIN, 0, stream>>>(x, src, dst, agg0, FIN);
    k_sage_gemm<<<NN / 16, 32, 0, stream>>>(agg0, deg, x, Wl0p, Wr0p, b0, h0, FIN);
    k_twin_gemm<<<NN / 16, 32, 0, stream>>>(x, Wr0p, h0t, FIN);

    // layer 1
    k_scatter<<<EE, HID, 0, stream>>>(h0, src, dst, agg1, HID);
    k_sage_gemm<<<NN / 16, 32, 0, stream>>>(agg1, deg, h0, Wl1p, Wr1p, b1, h1, HID);
    k_twin_gemm<<<NN / 16, 32, 0, stream>>>(h0t, Wr1p, h1t, HID);

    // attention + head
    k_attn<<<(NN + 7) / 8, 256, 0, stream>>>(h0, h0t, h1, h1t, hsum,
                                             out + (size_t)NN * CLS);
    k_head<<<NN / 16, 32, 0, stream>>>(hsum, outWp, outb, out);
}
